// InterSO3Conv_24206435681035
// MI455X (gfx1250) — matrix operand order repaired and checked by reading, not compiled, not run on hardware
//
#include <hip/hip_runtime.h>

// Problem constants (match setup_inputs: b=1, p=1024, c_in=32, c_out=64, na=60, ks=24, nn=32)
#define P_PTS 1024
#define C_IN  32
#define C_OUT 64
#define NA    60
#define KS    24
#define NN    32
#define CK    (C_IN * KS)          // 768
#define SIGMA_INV (1.0f / 0.16f)

typedef __bf16 bf16_t;
typedef __attribute__((ext_vector_type(16))) __bf16 v16bf;
typedef __attribute__((ext_vector_type(8)))  float  v8f;

// ---------------------------------------------------------------------------
// Kernel 1: rotated kernel points rk = anchors @ kernels^T, |rk|^2, and W->bf16
// ---------------------------------------------------------------------------
__global__ void __launch_bounds__(256)
prep_kernel(const float* __restrict__ W, const float* __restrict__ anchors,
            const float* __restrict__ kernels,
            float* __restrict__ rkx, float* __restrict__ rky,
            float* __restrict__ rkz, float* __restrict__ rk2,
            bf16_t* __restrict__ Wb) {
  const int t = blockIdx.x * 256 + threadIdx.x;
  if (t < NA * KS) {
    const int a = t / KS, k = t % KS;
    float r[3];
#pragma unroll
    for (int i = 0; i < 3; ++i)
      r[i] = anchors[a*9 + i*3 + 0] * kernels[k*3 + 0]
           + anchors[a*9 + i*3 + 1] * kernels[k*3 + 1]
           + anchors[a*9 + i*3 + 2] * kernels[k*3 + 2];
    rkx[t] = r[0]; rky[t] = r[1]; rkz[t] = r[2];
    rk2[t] = r[0]*r[0] + r[1]*r[1] + r[2]*r[2];
  }
  if (t < C_OUT * CK) Wb[t] = (bf16_t)W[t];
}

// ---------------------------------------------------------------------------
// Kernel 2: ball query (radius 1.0, first 32 indices in order, pad with first)
// xyz layout [3][1024]; outputs idx[p][32], rel coords + |rel|^2
// ---------------------------------------------------------------------------
__global__ void __launch_bounds__(256)
ballquery_kernel(const float* __restrict__ xyz,
                 int* __restrict__ idxarr,
                 float* __restrict__ gx, float* __restrict__ gy,
                 float* __restrict__ gz, float* __restrict__ g2) {
  __shared__ float sx[P_PTS], sy[P_PTS], sz[P_PTS];
  const int tid = threadIdx.x;
  for (int j = tid; j < P_PTS; j += 256) {
    sx[j] = xyz[j];
    sy[j] = xyz[P_PTS + j];
    sz[j] = xyz[2*P_PTS + j];
  }
  __syncthreads();
  const int q = blockIdx.x * 256 + tid;
  const float qx = sx[q], qy = sy[q], qz = sz[q];
  int cnt = 0, i0 = 0;
  float fx = 0.f, fy = 0.f, fz = 0.f;
  for (int j = 0; j < P_PTS; ++j) {
    const float dx = sx[j] - qx, dy = sy[j] - qy, dz = sz[j] - qz;
    const float d2 = dx*dx + dy*dy + dz*dz;
    if (d2 < 1.0f && cnt < NN) {
      if (cnt == 0) { i0 = j; fx = dx; fy = dy; fz = dz; }
      idxarr[q*NN + cnt] = j;
      gx[q*NN + cnt] = dx; gy[q*NN + cnt] = dy; gz[q*NN + cnt] = dz;
      g2[q*NN + cnt] = d2;
      ++cnt;
    }
  }
  const float f2 = fx*fx + fy*fy + fz*fz;
  for (int i = cnt; i < NN; ++i) {          // pad with first candidate (self always valid)
    idxarr[q*NN + i] = i0;
    gx[q*NN + i] = fx; gy[q*NN + i] = fy; gz[q*NN + i] = fz;
    g2[q*NN + i] = f2;
  }
}

// ---------------------------------------------------------------------------
// Kernel 3: fused kernel-correlation + SO3 conv, one workgroup per point.
//   stage2: NF[c*24+k][a] = sum_n gf[c][n][a] * w[a][k][n]   (WMMA K=32)
//   stage3: out[o][p][a]  = sum_ck W[o][ck] * NF[ck][a]      (WMMA K=768)
// LDS (bf16): sGF [32a][32c][32n]+pad, sW [32a][32k][32n], sNF [768][32a],
//             sWb [64][768]  -> ~280 KB total (<= 320 KB/workgroup per ISA)
// ---------------------------------------------------------------------------
#define GF_STRIDE 1042   // padded per-anchor stride (elements) to spread LDS banks
#define SGF_ELEMS (32 * GF_STRIDE)
#define SW_ELEMS  (32 * 32 * 32)
#define SNF_ELEMS (CK * 32)
#define SWB_ELEMS (C_OUT * CK)
#define SMEM_BYTES ((SGF_ELEMS + SW_ELEMS + SNF_ELEMS + SWB_ELEMS) * 2 + 32*4 + 96*4 + 32*4)

__global__ void __launch_bounds__(256)
so3conv_kernel(const float* __restrict__ feats,
               const int*   __restrict__ idxarr,
               const float* __restrict__ gxarr, const float* __restrict__ gyarr,
               const float* __restrict__ gzarr, const float* __restrict__ g2arr,
               const float* __restrict__ rkx, const float* __restrict__ rky,
               const float* __restrict__ rkz, const float* __restrict__ rk2,
               const bf16_t* __restrict__ Wb,
               float* __restrict__ out) {
  const int p    = blockIdx.x;
  const int tid  = threadIdx.x;
  const int lane = tid & 31;
  const int wave = tid >> 5;
  const int lo16 = lane & 15;
  const int hi   = lane >> 4;

  extern __shared__ char smem_raw[];
  bf16_t* sGF = (bf16_t*)smem_raw;              // [32][GF_STRIDE]
  bf16_t* sW  = sGF + SGF_ELEMS;                // [32][32][32]
  bf16_t* sNF = sW  + SW_ELEMS;                 // [768][32]
  bf16_t* sWb = sNF + SNF_ELEMS;                // [64][768]
  int*   sIdx = (int*)(sWb + SWB_ELEMS);        // [32]
  float* sGX  = (float*)(sIdx + 32);            // [3][32]
  float* sG2  = sGX + 96;                       // [32]

  // one-time: stage bf16 W into LDS (dword-wide coalesced copy)
  {
    const unsigned int* src = (const unsigned int*)Wb;
    unsigned int* dst = (unsigned int*)sWb;
    for (int e = tid; e < SWB_ELEMS / 2; e += 256) dst[e] = src[e];
  }
  if (tid < NN) {
    sIdx[tid]     = idxarr[p*NN + tid];
    sGX[tid]      = gxarr[p*NN + tid];
    sGX[32 + tid] = gyarr[p*NN + tid];
    sGX[64 + tid] = gzarr[p*NN + tid];
    sG2[tid]      = g2arr[p*NN + tid];
  }
  __syncthreads();

  for (int chunk = 0; chunk < 2; ++chunk) {
    const int abase = chunk * 32;

    // ---- phase 1a: gather neighbor features (coalesced along a) ----
    for (int e = tid; e < 32*32*32; e += 256) {
      const int al = e & 31;
      const int n  = (e >> 5) & 31;
      const int c  = e >> 10;
      const int a  = abase + al;
      float v = 0.f;
      if (a < NA) v = feats[(c*P_PTS + sIdx[n])*NA + a];
      sGF[al*GF_STRIDE + c*32 + n] = (bf16_t)v;
    }
    // ---- phase 1b: interpolation weights w = relu(1 - d2/sigma), k padded to 32 ----
    for (int e = tid; e < 32*32*32; e += 256) {
      const int n  = e & 31;
      const int k  = (e >> 5) & 31;
      const int al = e >> 10;
      const int a  = abase + al;
      float wv = 0.f;
      if (a < NA && k < KS) {
        const int ak = a*KS + k;
        const float d2 = sG2[n] + rk2[ak]
          - 2.f*(rkx[ak]*sGX[n] + rky[ak]*sGX[32+n] + rkz[ak]*sGX[64+n]);
        wv = 1.f - d2 * SIGMA_INV;
        wv = wv > 0.f ? wv : 0.f;
      }
      sW[e] = (bf16_t)wv;
    }
    __syncthreads();

    // ---- phase 2: per-anchor (32c x 24k, K=32n) GEMM via WMMA -> sNF ----
    for (int j = 0; j < 4; ++j) {
      const int al = wave*4 + j;
      if (abase + al >= NA) continue;          // wave-uniform skip
      for (int ct = 0; ct < 2; ++ct) {
        const int c = ct*16 + lo16;
        for (int kt = 0; kt < 2; ++kt) {
          const int k = kt*16 + lo16;
          v16bf av, bv;
#pragma unroll
          for (int i = 0; i < 16; ++i) {
            // A (16x32 bf16): lanes 0-15 K=0..15, lanes 16-31 K=16..31
            const int na_ = hi*16 + i;
            av[i] = sGF[al*GF_STRIDE + c*32 + na_];
            // B (32x16 bf16): lanes 0-15 K=0..15, lanes 16-31 K=16..31; B[n][k]=w[k][n]
            bv[i] = sW[(al*32 + k)*32 + (hi*16 + i)];
          }
          v8f acc = {};
          acc = __builtin_amdgcn_wmma_f32_16x16x32_bf16(
                  false, av, false, bv, (short)0, acc, false, false);
          if (k < KS) {
#pragma unroll
            for (int r = 0; r < 8; ++r) {
              const int c2 = ct*16 + hi*8 + r;       // C/D: M = r + 8*(lane>=16)
              sNF[(c2*KS + k)*32 + al] = (bf16_t)acc[r];
            }
          }
        }
      }
    }
    __syncthreads();

    // ---- phase 3: out tile (64o x 32a) = W(64x768) @ NF(768x32) ----
    {
      const int mt = wave >> 1;                // o-tile 0..3
      const int nt = wave & 1;                 // a-tile 0..1
      const int o  = mt*16 + lo16;
      const int al = nt*16 + lo16;
      v8f acc = {};
      for (int kk = 0; kk < CK/32; ++kk) {     // 24 chained WMMA, K=768
        v16bf av, bv;
#pragma unroll
        for (int i = 0; i < 16; ++i) {
          const int cka = kk*32 + hi*16 + i;
          av[i] = sWb[o*CK + cka];
          const int ckb = kk*32 + hi*16 + i;
          bv[i] = sNF[ckb*32 + al];
        }
        acc = __builtin_amdgcn_wmma_f32_16x16x32_bf16(
                false, av, false, bv, (short)0, acc, false, false);
      }
      const int a = abase + al;
      if (a < NA) {
#pragma unroll
        for (int r = 0; r < 8; ++r) {
          const int o2 = mt*16 + hi*8 + r;
          out[(o2*P_PTS + p)*NA + a] = acc[r];
        }
      }
    }
    __syncthreads();
  }
}

// ---------------------------------------------------------------------------
extern "C" void kernel_launch(void* const* d_in, const int* in_sizes, int n_in,
                              void* d_out, int out_size, void* d_ws, size_t ws_size,
                              hipStream_t stream) {
  (void)in_sizes; (void)n_in; (void)out_size; (void)ws_size;
  const float* xyz     = (const float*)d_in[0];   // [3][1024]
  const float* feats   = (const float*)d_in[1];   // [32][1024][60]
  const float* W       = (const float*)d_in[2];   // [64][768]
  const float* anchors = (const float*)d_in[3];   // [60][3][3]
  const float* kernels = (const float*)d_in[4];   // [24][3]
  float* out = (float*)d_out;                     // [64][1024][60]

  char* ws = (char*)d_ws;
  int*   idxarr = (int*)ws;          ws += (size_t)P_PTS*NN*4;   // 128 KB
  float* gx     = (float*)ws;        ws += (size_t)P_PTS*NN*4;
  float* gy     = (float*)ws;        ws += (size_t)P_PTS*NN*4;
  float* gz     = (float*)ws;        ws += (size_t)P_PTS*NN*4;
  float* g2     = (float*)ws;        ws += (size_t)P_PTS*NN*4;
  float* rkx    = (float*)ws;        ws += 6144;                 // 60*24*4 rounded
  float* rky    = (float*)ws;        ws += 6144;
  float* rkz    = (float*)ws;        ws += 6144;
  float* rk2    = (float*)ws;        ws += 6144;
  bf16_t* Wb    = (bf16_t*)ws;                                   // 96 KB bf16

  prep_kernel<<<(C_OUT*CK + 255)/256, 256, 0, stream>>>(W, anchors, kernels,
                                                        rkx, rky, rkz, rk2, Wb);
  ballquery_kernel<<<P_PTS/256, 256, 0, stream>>>(xyz, idxarr, gx, gy, gz, g2);
  so3conv_kernel<<<P_PTS, 256, SMEM_BYTES, stream>>>(feats, idxarr, gx, gy, gz, g2,
                                                     rkx, rky, rkz, rk2, Wb, out);
}
